// MemoryEfficientAttention_56916906606632
// MI455X (gfx1250) — matrix operand order repaired and checked by reading, not compiled, not run on hardware
//
#include <hip/hip_runtime.h>
#include <hip/hip_bf16.h>

// ---------------------------------------------------------------------------
// bf16 WMMA attention for MI455X (gfx1250, wave32).
// - All heavy math through v_wmma_f32_16x16x32_bf16 (fp32 accumulation).
// - GEMMs: LDS-staged, double-buffered via gfx1250 async global->LDS copies
//   (ASYNCcnt); fragment loads batched ahead of the WMMA chain so waits are
//   graduated instead of full stalls.
// - Attention: flash-style, online softmax, batched fragment loads,
//   global_prefetch for the next key block.
// ---------------------------------------------------------------------------

typedef __attribute__((ext_vector_type(16))) __bf16 v16bf;
typedef __attribute__((ext_vector_type(8)))  float  v8f;

union FragB16 { v16bf v; uint4 q[2]; };

// A-matrix 16x32 (M x K): lanes 0-15 -> M=lane, K runs {0..7,16..23};
// lanes 16-31 -> M=lane-16, K runs {8..15,24..31}.
__device__ __forceinline__ v16bf load_a_frag(const __bf16* rowPtr, int half) {
    FragB16 f;
    f.q[0] = *reinterpret_cast<const uint4*>(rowPtr + half * 8);
    f.q[1] = *reinterpret_cast<const uint4*>(rowPtr + 16 + half * 8);
    return f.v;
}

// B-matrix 32x16 (K x N): lane holds column N=lane&15, K runs {half*8..+7, 16+half*8..+7}.
__device__ __forceinline__ v16bf load_b_frag(const __bf16* colPtr, int half) {
    FragB16 f;
    const __bf16* p = colPtr + half * 8;
    f.q[0] = *reinterpret_cast<const uint4*>(p);
    f.q[1] = *reinterpret_cast<const uint4*>(p + 16);
    return f.v;
}

// ---------------------------------------------------------------------------
// Async global->LDS (CDNA5, ASYNCcnt).
// ---------------------------------------------------------------------------
#if __has_builtin(__builtin_amdgcn_global_load_async_to_lds_b128)
#define HAVE_ASYNC_LDS 1
typedef int async_v4i __attribute__((vector_size(16)));
__device__ __forceinline__ void async16(const __bf16* g, __bf16* l) {
    __builtin_amdgcn_global_load_async_to_lds_b128(
        (__attribute__((address_space(1))) async_v4i*)(unsigned long long)(g),
        (__attribute__((address_space(3))) async_v4i*)(unsigned int)(unsigned long long)(l),
        0, 0);
}
#else
#define HAVE_ASYNC_LDS 0
#endif

struct CopyRegs { uint4 a0, a1, b0, b1; };

#define LDS_STRIDE 40   // 32 payload + 8 pad bf16 -> spreads LDS banks

__device__ __forceinline__ void tile_copy_start(const __bf16* gA, const __bf16* gB,
                                                int K, __bf16* la, __bf16* lb,
                                                CopyRegs& r) {
#if HAVE_ASYNC_LDS
    async16(gA, la);
    async16(gA + (size_t)64 * K, la + 64 * LDS_STRIDE);
    async16(gB, lb);
    async16(gB + (size_t)64 * K, lb + 64 * LDS_STRIDE);
    (void)r;
#else
    r.a0 = *reinterpret_cast<const uint4*>(gA);
    r.a1 = *reinterpret_cast<const uint4*>(gA + (size_t)64 * K);
    r.b0 = *reinterpret_cast<const uint4*>(gB);
    r.b1 = *reinterpret_cast<const uint4*>(gB + (size_t)64 * K);
#endif
}

__device__ __forceinline__ void tile_copy_finish(__bf16* la, __bf16* lb,
                                                 const CopyRegs& r) {
#if HAVE_ASYNC_LDS
    asm volatile("s_wait_asynccnt 0" ::: "memory");
    (void)la; (void)lb; (void)r;
#else
    *reinterpret_cast<uint4*>(la) = r.a0;
    *reinterpret_cast<uint4*>(la + 64 * LDS_STRIDE) = r.a1;
    *reinterpret_cast<uint4*>(lb) = r.b0;
    *reinterpret_cast<uint4*>(lb + 64 * LDS_STRIDE) = r.b1;
#endif
}

__global__ void f32_to_bf16_kernel(const float* __restrict__ in,
                                   __bf16* __restrict__ out, int n) {
    int i = blockIdx.x * blockDim.x + threadIdx.x;
    if (i < n) out[i] = (__bf16)in[i];
}

// WT[n*K + k] = W[k*N + n]
__global__ void transpose_f32_to_bf16(const float* __restrict__ W,
                                      __bf16* __restrict__ WT, int K, int N) {
    int idx = blockIdx.x * blockDim.x + threadIdx.x;
    if (idx >= K * N) return;
    int k = idx / N, n = idx % N;
    WT[(size_t)n * K + k] = (__bf16)W[idx];
}

// C = A * B (+bias). A:[M][K] bf16 row-major, BT:[N][K] bf16.
// 256 threads = 8 waves; 128x128 block tile; K-step 32, LDS double-buffered.
// mode 0: bf16 row-major; mode 1: bf16 transposed (ldT rows); mode 2: fp32.
__global__ __launch_bounds__(256) void gemm_bf16_wmma(
    const __bf16* __restrict__ A, const __bf16* __restrict__ BT,
    const float* __restrict__ bias, void* __restrict__ C,
    int M, int N, int K, int mode, int ldT)
{
    __shared__ __bf16 ldsA[2][128 * LDS_STRIDE];
    __shared__ __bf16 ldsB[2][128 * LDS_STRIDE];

    const int tid  = threadIdx.x;
    const int lane = tid & 31, wave = tid >> 5;
    const int half = lane >> 4, l16 = lane & 15;
    const int rowBase = blockIdx.x * 128 + wave * 16;
    const int colBase = blockIdx.y * 128;

    const int cr0 = tid >> 2;            // row 0..63 (+64 for second chunk)
    const int cc0 = (tid & 3) * 8;       // col 0,8,16,24
    const __bf16* gA = A + (size_t)(blockIdx.x * 128 + cr0) * K + cc0;
    const __bf16* gB = BT + (size_t)(colBase + cr0) * K + cc0;
    const int ldsOff = cr0 * LDS_STRIDE + cc0;

    CopyRegs cr;
    tile_copy_start(gA, gB, K, &ldsA[0][ldsOff], &ldsB[0][ldsOff], cr);
    tile_copy_finish(&ldsA[0][ldsOff], &ldsB[0][ldsOff], cr);
    __syncthreads();

    v8f acc[8] = {};
    const int nk = K / 32;
    for (int kt = 0; kt < nk; ++kt) {
        const int buf = kt & 1, nbuf = buf ^ 1;
        if (kt + 1 < nk)
            tile_copy_start(gA + (kt + 1) * 32, gB + (kt + 1) * 32, K,
                            &ldsA[nbuf][ldsOff], &ldsB[nbuf][ldsOff], cr);

        // Batch all fragment loads ahead of the WMMA chain so the in-order
        // LDS returns allow graduated s_wait_dscnt instead of full stalls.
        v16bf af = load_a_frag(&ldsA[buf][(wave * 16 + l16) * LDS_STRIDE], half);
        v16bf bfrag[8];
#pragma unroll
        for (int t = 0; t < 8; ++t)
            bfrag[t] = load_b_frag(&ldsB[buf][(t * 16 + l16) * LDS_STRIDE], half);
#pragma unroll
        for (int t = 0; t < 8; ++t)
            acc[t] = __builtin_amdgcn_wmma_f32_16x16x32_bf16(
                false, af, false, bfrag[t], (short)0, acc[t], false, false);

        if (kt + 1 < nk)
            tile_copy_finish(&ldsA[nbuf][ldsOff], &ldsB[nbuf][ldsOff], cr);
        __syncthreads();
    }

#pragma unroll
    for (int t = 0; t < 8; ++t) {
        int n = colBase + t * 16 + l16;
        float bvv = bias[n];
#pragma unroll
        for (int r = 0; r < 8; ++r) {
            int m = rowBase + r + half * 8;
            float v = acc[t][r] + bvv;
            if (mode == 0)      ((__bf16*)C)[(size_t)m * N + n]   = (__bf16)v;
            else if (mode == 1) ((__bf16*)C)[(size_t)n * ldT + m] = (__bf16)v;
            else                ((float*)C)[(size_t)m * N + n]    = v;
        }
    }
}

#define S_LEN  4096
#define DMODEL 1024

// Flash attention, full d=1024. One workgroup (8 waves) per 16-query block.
__global__ __launch_bounds__(256) void flash_attn_wmma(
    const __bf16* __restrict__ Qg, const __bf16* __restrict__ Kg,
    const __bf16* __restrict__ VT, __bf16* __restrict__ O)
{
    __shared__ __bf16 Qs[16 * DMODEL];   // 32 KB
    __shared__ float  Sbuf[16 * 32];
    __shared__ __bf16 Pbuf[16 * 32];
    __shared__ float  mrow[16], lrow[16], crow[16];

    const int tid  = threadIdx.x;
    const int lane = tid & 31, wave = tid >> 5;
    const int half = lane >> 4, l16 = lane & 15;
    const int batch = blockIdx.y;
    const int qbase = blockIdx.x * 16;
    const int dbase = wave * 128;

    const uint4* qsrc = reinterpret_cast<const uint4*>(
        Qg + ((size_t)batch * S_LEN + qbase) * DMODEL);
    uint4* qdst = reinterpret_cast<uint4*>(Qs);
#pragma unroll
    for (int i = 0; i < 8; ++i) qdst[tid + i * 256] = qsrc[tid + i * 256];

    if (tid < 16) { mrow[tid] = -3.0e38f; lrow[tid] = 0.0f; }

    v8f oacc[8] = {};
    __syncthreads();

    for (int j = 0; j < S_LEN / 32; ++j) {
        if (j + 1 < S_LEN / 32) {
            const __bf16* kpn =
                Kg + ((size_t)batch * S_LEN + (j + 1) * 32 + l16) * DMODEL + dbase;
            __builtin_prefetch(kpn, 0, 1);
            __builtin_prefetch(kpn + 16 * DMODEL, 0, 1);
            const __bf16* vpn =
                VT + ((size_t)batch * DMODEL + dbase + l16) * S_LEN + (j + 1) * 32;
            __builtin_prefetch(vpn, 0, 1);
        }

        Sbuf[tid] = 0.0f; Sbuf[tid + 256] = 0.0f;
        __syncthreads();

        // Batch the 8 global K-fragment loads, then run the WMMA chain.
        v16bf kb0[4], kb1[4];
#pragma unroll
        for (int kk = 0; kk < 4; ++kk) {
            const __bf16* kp0 =
                Kg + ((size_t)batch * S_LEN + j * 32 + l16) * DMODEL + dbase + kk * 32;
            kb0[kk] = load_b_frag(kp0, half);
            kb1[kk] = load_b_frag(kp0 + 16 * DMODEL, half);
        }
        v8f s0 = {}, s1 = {};
#pragma unroll
        for (int kk = 0; kk < 4; ++kk) {
            v16bf af = load_a_frag(Qs + l16 * DMODEL + dbase + kk * 32, half);
            s0 = __builtin_amdgcn_wmma_f32_16x16x32_bf16(
                false, af, false, kb0[kk], (short)0, s0, false, false);
            s1 = __builtin_amdgcn_wmma_f32_16x16x32_bf16(
                false, af, false, kb1[kk], (short)0, s1, false, false);
        }
#pragma unroll
        for (int r = 0; r < 8; ++r) {
            int m = r + half * 8;
            atomicAdd(&Sbuf[m * 32 + l16],      s0[r]);
            atomicAdd(&Sbuf[m * 32 + 16 + l16], s1[r]);
        }
        __syncthreads();

        // Online softmax (scale = 1/sqrt(64) = 0.125), one thread per row.
        if (tid < 16) {
            float ps[32];
            float mold = mrow[tid];
            float mnew = mold;
#pragma unroll
            for (int c = 0; c < 32; ++c) {
                ps[c] = Sbuf[tid * 32 + c] * 0.125f;
                mnew = fmaxf(mnew, ps[c]);
            }
            float corr = __expf(mold - mnew);
            float sum = 0.0f;
#pragma unroll
            for (int c = 0; c < 32; ++c) {
                float p = __expf(ps[c] - mnew);
                sum += p;
                Pbuf[tid * 32 + c] = (__bf16)p;
            }
            mrow[tid] = mnew;
            crow[tid] = corr;
            lrow[tid] = lrow[tid] * corr + sum;
        }
        __syncthreads();

        // Batch the 8 V-fragment loads; overlap the O-rescale VALU work with
        // their latency, then run the PV WMMA chain.
        v16bf vfrag[8];
#pragma unroll
        for (int t = 0; t < 8; ++t) {
            const __bf16* vp =
                VT + ((size_t)batch * DMODEL + dbase + t * 16 + l16) * S_LEN + j * 32;
            vfrag[t] = load_b_frag(vp, half);
        }
        float cs[8];
#pragma unroll
        for (int r = 0; r < 8; ++r) cs[r] = crow[r + half * 8];
#pragma unroll
        for (int t = 0; t < 8; ++t)
#pragma unroll
            for (int r = 0; r < 8; ++r) oacc[t][r] *= cs[r];
        v16bf pf = load_a_frag(Pbuf + l16 * 32, half);
#pragma unroll
        for (int t = 0; t < 8; ++t)
            oacc[t] = __builtin_amdgcn_wmma_f32_16x16x32_bf16(
                false, pf, false, vfrag[t], (short)0, oacc[t], false, false);
        __syncthreads();
    }

    float li[8];
#pragma unroll
    for (int r = 0; r < 8; ++r) li[r] = 1.0f / lrow[r + half * 8];
#pragma unroll
    for (int t = 0; t < 8; ++t) {
#pragma unroll
        for (int r = 0; r < 8; ++r) {
            int m = qbase + r + half * 8;
            int n = dbase + t * 16 + l16;
            O[((size_t)batch * S_LEN + m) * DMODEL + n] =
                (__bf16)(oacc[t][r] * li[r]);
        }
    }
}

extern "C" void kernel_launch(void* const* d_in, const int* in_sizes, int n_in,
                              void* d_out, int out_size, void* d_ws, size_t ws_size,
                              hipStream_t stream) {
    const float* x  = (const float*)d_in[0];
    const float* Wq = (const float*)d_in[1];
    const float* bq = (const float*)d_in[2];
    const float* Wk = (const float*)d_in[3];
    const float* bk = (const float*)d_in[4];
    const float* Wv = (const float*)d_in[5];
    const float* bv = (const float*)d_in[6];
    const float* Wo = (const float*)d_in[7];
    const float* bo = (const float*)d_in[8];
    float* out = (float*)d_out;

    const size_t S = 4096, D = 1024, MD = 2 * S;

    __bf16* ws   = (__bf16*)d_ws;
    __bf16* xb   = ws; ws += MD * D;
    __bf16* qb   = ws; ws += MD * D;
    __bf16* kbuf = ws; ws += MD * D;
    __bf16* vbT  = ws; ws += MD * D;   // per batch: [D][S]
    __bf16* obuf = ws; ws += MD * D;
    __bf16* WqT  = ws; ws += D * D;
    __bf16* WkT  = ws; ws += D * D;
    __bf16* WvT  = ws; ws += D * D;
    __bf16* WoT  = ws; ws += D * D;
    (void)ws_size; (void)in_sizes; (void)n_in; (void)out_size;

    f32_to_bf16_kernel<<<dim3((unsigned)((MD * D) / 256)), 256, 0, stream>>>(
        x, xb, (int)(MD * D));
    transpose_f32_to_bf16<<<dim3((unsigned)((D * D) / 256)), 256, 0, stream>>>(
        Wq, WqT, (int)D, (int)D);
    transpose_f32_to_bf16<<<dim3((unsigned)((D * D) / 256)), 256, 0, stream>>>(
        Wk, WkT, (int)D, (int)D);
    transpose_f32_to_bf16<<<dim3((unsigned)((D * D) / 256)), 256, 0, stream>>>(
        Wv, WvT, (int)D, (int)D);
    transpose_f32_to_bf16<<<dim3((unsigned)((D * D) / 256)), 256, 0, stream>>>(
        Wo, WoT, (int)D, (int)D);

    dim3 blk(256);
    dim3 gFull((unsigned)(MD / 128), (unsigned)(D / 128));
    dim3 gHalf((unsigned)(S / 128),  (unsigned)(D / 128));

    gemm_bf16_wmma<<<gFull, blk, 0, stream>>>(xb, WqT, bq, qb,
                                              (int)MD, (int)D, (int)D, 0, 0);
    gemm_bf16_wmma<<<gFull, blk, 0, stream>>>(xb, WkT, bk, kbuf,
                                              (int)MD, (int)D, (int)D, 0, 0);
    for (int b = 0; b < 2; ++b)
        gemm_bf16_wmma<<<gHalf, blk, 0, stream>>>(
            xb + (size_t)b * S * D, WvT, bv, vbT + (size_t)b * D * S,
            (int)S, (int)D, (int)D, 1, (int)S);

    flash_attn_wmma<<<dim3((unsigned)(S / 16), 2), blk, 0, stream>>>(
        qb, kbuf, vbT, obuf);

    gemm_bf16_wmma<<<gFull, blk, 0, stream>>>(obuf, WoT, bo, out,
                                              (int)MD, (int)D, (int)D, 2, 0);
}
